// LSTMLayer_30356828848340
// MI455X (gfx1250) — matrix operand order repaired and checked by reading, not compiled, not run on hardware
//
#include <hip/hip_runtime.h>
#include <hip/hip_bf16.h>

// LSTM layer: T=2048, B=64, I=256, H=256.
// Phase 1: XG = X @ W_ih^T + (b_ih + b_hh)   (bf16 WMMA, fp32 accum) -> ws
// Phase 2: persistent scan kernel (16 WGs):
//   - W_hh slice preloaded via TDM (tensor_load_to_lds) + LDS convert
//   - h_{t-1} staged each step with GLOBAL_LOAD_ASYNC_TO_LDS_B128 from a
//     bf16 ping-pong buffer in ws (written by the previous step's pointwise)
//   - one grid barrier per timestep, cell state in registers.

#define T_STEPS 2048
#define BATCH   64
#define IDIM    256
#define HDIM    256
#define GDIM    1024   // 4*H

typedef __attribute__((ext_vector_type(16))) __bf16 v16bf;
typedef __attribute__((ext_vector_type(8)))  __bf16 v8bf;
typedef __attribute__((ext_vector_type(8)))  float  v8f;
typedef __attribute__((ext_vector_type(4)))  float  f32x4;
typedef __attribute__((ext_vector_type(4)))  unsigned u32x4;
typedef __attribute__((ext_vector_type(8)))  unsigned u32x8;

// Scan-kernel LDS layout (dynamic LDS, no static LDS in that kernel => base 0)
#define LDS_WS_OFF    0u        // Ws: W_hh slice, bf16 [64 cols][256 k] (32 KB)
#define LDS_HS_OFF    32768u    // Hs: h_{t-1}, bf16 [64 rows][256 k]    (32 KB)
#define LDS_STAGE_OFF 65536u    // fp32 TDM staging [64][256]            (64 KB)
#define LDS_TOTAL     (128 * 1024)

static __device__ __forceinline__ v8f wmma_bf16(v16bf a, v16bf b, v8f c) {
  // D = A(16x32 bf16) * B(32x16 bf16) + C(16x16 f32)
  return __builtin_amdgcn_wmma_f32_16x16x32_bf16(
      /*neg_a=*/false, a, /*neg_b=*/false, b,
      /*c_mod=*/(short)0, c, /*reuse_a=*/false, /*reuse_b=*/false);
}

static __device__ __forceinline__ float sigmoidf_fast(float x) {
  return 1.0f / (1.0f + __expf(-x));
}

static __device__ __forceinline__ void cvt_store16(__bf16* dst, const float* src) {
  #pragma unroll
  for (int v = 0; v < 4; ++v) {
    f32x4 d = *(const f32x4*)(src + v * 4);
    dst[v * 4 + 0] = (__bf16)d[0];
    dst[v * 4 + 1] = (__bf16)d[1];
    dst[v * 4 + 2] = (__bf16)d[2];
    dst[v * 4 + 3] = (__bf16)d[3];
  }
}

// ---------------------------------------------------------------------------
// Kernel A: XG[m, n] = sum_k X[m,k] * W_ih[n,k] + b_ih[n] + b_hh[n]
// M = T*B = 131072, N = 1024, K = 256. Block tile 128x128, 8 waves.
// ---------------------------------------------------------------------------
__global__ __launch_bounds__(256)
void xgates_gemm(const float* __restrict__ x,     // [T*B, I]
                 const float* __restrict__ Wih,   // [4H, I]
                 const float* __restrict__ b_ih,  // [4H]
                 const float* __restrict__ b_hh,  // [4H]
                 float* __restrict__ xg)          // [T*B, 4H]
{
  __shared__ __bf16 As[128 * 32];  // A tile, row-major [row][k]
  __shared__ __bf16 Bs[128 * 32];  // B tile, [col][k] (== W_ih row-major slice)

  const int mbase = blockIdx.x * 128;
  const int nbase = blockIdx.y * 128;
  const int tid   = threadIdx.x;
  const int lane  = tid & 31;
  const int w     = tid >> 5;           // wave 0..7
  const int mw    = (w & 3) * 32;       // wave row offset (4 waves down)
  const int nw    = (w >> 2) * 64;      // wave col offset (2 waves across)
  const int l16   = lane & 15;
  const int hi    = lane >> 4;          // 0: lanes 0-15, 1: lanes 16-31

  v8f acc[2][4];
  #pragma unroll
  for (int i = 0; i < 2; ++i)
    #pragma unroll
    for (int j = 0; j < 4; ++j)
      #pragma unroll
      for (int e = 0; e < 8; ++e) acc[i][j][e] = 0.0f;

  const int lrow  = tid >> 1;           // 0..127 (tile row / tile col)
  const int khalf = (tid & 1) * 16;     // 0 or 16

  for (int kb = 0; kb < 8; ++kb) {
    const int k0 = kb * 32;
    cvt_store16(As + lrow * 32 + khalf,
                x + (size_t)(mbase + lrow) * IDIM + k0 + khalf);
    cvt_store16(Bs + lrow * 32 + khalf,
                Wih + (size_t)(nbase + lrow) * IDIM + k0 + khalf);
    __syncthreads();

    #pragma unroll
    for (int mt = 0; mt < 2; ++mt) {
      // A fragment: lanes 0-15 K 0-15, lanes 16-31 K 16-31,
      // row = M = lane%16.
      const __bf16* ap = As + (mw + mt * 16 + l16) * 32 + hi * 16;
      v8bf alo = *(const v8bf*)(ap);
      v8bf ahi = *(const v8bf*)(ap + 8);
      v16bf afrag = __builtin_shufflevector(alo, ahi,
          0, 1, 2, 3, 4, 5, 6, 7, 8, 9, 10, 11, 12, 13, 14, 15);
      #pragma unroll
      for (int nt = 0; nt < 4; ++nt) {
        // B fragment: lanes 0-15 K 0-15, lanes 16-31 K 16-31, col = lane%16
        const __bf16* bp = Bs + (nw + nt * 16 + l16) * 32 + hi * 16;
        v16bf bfrag = *(const v16bf*)(bp);
        acc[mt][nt] = wmma_bf16(afrag, bfrag, acc[mt][nt]);
      }
    }
    __syncthreads();
  }

  #pragma unroll
  for (int mt = 0; mt < 2; ++mt)
    #pragma unroll
    for (int nt = 0; nt < 4; ++nt) {
      const int n = nbase + nw + nt * 16 + l16;
      const float bias = b_ih[n] + b_hh[n];
      #pragma unroll
      for (int r = 0; r < 8; ++r) {
        const int m = mbase + mw + mt * 16 + r + hi * 8;
        xg[(size_t)m * GDIM + n] = acc[mt][nt][r] + bias;
      }
    }
}

// ---------------------------------------------------------------------------
// Kernel B: persistent recurrent scan. 16 WGs x 128 threads (4 waves).
// ---------------------------------------------------------------------------
__global__ __launch_bounds__(128)
void lstm_scan(const float* __restrict__ h0,     // [B, H]
               const float* __restrict__ c0,     // [B, H]
               const float* __restrict__ Whh,    // [4H, H]
               const float* __restrict__ xg,     // [T*B, 4H]
               float* __restrict__ out,          // outputs [T,B,H] | hT | cT
               __bf16* __restrict__ hb,          // bf16 h ping-pong, 2x[B,H]
               unsigned int* __restrict__ counter)
{
  extern __shared__ char smem_raw[];
  __bf16* Ws    = (__bf16*)(smem_raw + LDS_WS_OFF);    // [64 cols][256 k]
  __bf16* Hs    = (__bf16*)(smem_raw + LDS_HS_OFF);    // [64 rows][256 k]
  float*  Wstg  = (float*)(smem_raw + LDS_STAGE_OFF);  // [64][256] fp32

  const int wg    = blockIdx.x;        // 0..15
  const int hbase = wg * 16;           // h-column base
  const int tid   = threadIdx.x;
  const int lane  = tid & 31;
  const int w     = tid >> 5;          // 0..3 = batch tile (M)
  const int l16   = lane & 15;
  const int hi    = lane >> 4;

  // ---- One-time W_hh preload via Tensor Data Mover -----------------------
  // 4 descriptors, one per gate: 2D tile 256(k) x 16(rows) fp32 from global
  // rows [g*256+hbase, +16), row stride 256 elements, into LDS fp32 staging.
  if (w == 0) {
    #pragma unroll
    for (int g = 0; g < 4; ++g) {
      const unsigned long long ga =
          (unsigned long long)(uintptr_t)(Whh + (size_t)(g * 256 + hbase) * HDIM);
      // D# group 0: count=1 | lds_addr | global_addr[56:0] | type=2
      u32x4 d0;
      d0[0] = 1u;                                        // count=1, user mode
      d0[1] = LDS_STAGE_OFF + (unsigned)g * 16384u;      // lds_addr (bytes)
      d0[2] = (unsigned)(ga & 0xFFFFFFFFull);            // global_addr[31:0]
      d0[3] = (unsigned)((ga >> 32) & 0x1FFFFFFull) | (2u << 30); // [56:32]|type
      // D# group 1: data_size=4B; tensor 256x16; tile 256x16; dim0 stride 256
      u32x8 d1;
      d1[0] = 2u << 16;      // workgroup_mask=0, data_size=2 (4 bytes)
      d1[1] = 256u << 16;    // tensor_dim0[15:0] in bits[31:16]
      d1[2] = 16u << 16;     // tensor_dim0 hi=0, tensor_dim1[15:0]=16
      d1[3] = 256u << 16;    // tensor_dim1 hi=0, tile_dim0=256
      d1[4] = 16u;           // tile_dim1=16, tile_dim2=0
      d1[5] = 256u;          // tensor_dim0_stride[31:0] = 256 elements
      d1[6] = 0u;            // stride hi, tensor_dim1_stride lo (unused)
      d1[7] = 0u;
      asm volatile("tensor_load_to_lds %0, %1" :: "s"(d0), "s"(d1) : "memory");
    }
    asm volatile("s_wait_tensorcnt 0x0" ::: "memory");
  }
  __syncthreads();
  // LDS fp32 -> Ws bf16 (layout already [lc = g*16+row][k])
  {
    const int base = tid * 128;                 // 16384 elems / 128 threads
    #pragma unroll
    for (int v = 0; v < 8; ++v)
      cvt_store16(Ws + base + v * 16, Wstg + base + v * 16);
  }

  // Cell-state fragment (16x16), lives in registers for the whole scan
  v8f cfrag;
  #pragma unroll
  for (int r = 0; r < 8; ++r) {
    const int m = w * 16 + r + hi * 8;
    cfrag[r] = c0[(size_t)m * HDIM + hbase + l16];
  }
  __syncthreads();

  for (int t = 0; t < T_STEPS; ++t) {
    // ---- Stage h_{t-1} into Hs ----
    if (t == 0) {
      // fp32 h0 -> bf16 LDS (VALU convert, once)
      const int row = tid >> 1;
      const int kh  = (tid & 1) * 128;
      const float* src = h0 + (size_t)row * HDIM + kh;
      __bf16* dst = Hs + row * 256 + kh;
      #pragma unroll
      for (int v = 0; v < 8; ++v) cvt_store16(dst + v * 16, src + v * 16);
    } else {
      // Raw bf16 async copy: 32 KB, 128 threads x 16 x b128 (ASYNCcnt)
      const __bf16* src = hb + (size_t)((t - 1) & 1) * (BATCH * HDIM);
      unsigned lds_addr = LDS_HS_OFF + (unsigned)tid * 16u;
      unsigned long long gaddr =
          (unsigned long long)(uintptr_t)src + (unsigned long long)tid * 16u;
      #pragma unroll
      for (int i = 0; i < 16; ++i) {
        asm volatile("global_load_async_to_lds_b128 %0, %1, off"
                     :: "v"(lds_addr), "v"(gaddr) : "memory");
        lds_addr += 2048u;
        gaddr    += 2048ull;
      }
    }

    // ---- Accumulators init with precomputed x-gate contribution (fp32) ----
    v8f acc[4];
    #pragma unroll
    for (int g = 0; g < 4; ++g) {
      const int gcol = g * 256 + hbase + l16;
      #pragma unroll
      for (int r = 0; r < 8; ++r) {
        const int m = w * 16 + r + hi * 8;
        acc[g][r] = xg[(size_t)(t * BATCH + m) * GDIM + gcol];
      }
    }

    if (t != 0) asm volatile("s_wait_asynccnt 0x0" ::: "memory");
    __syncthreads();

    // ---- gates += h_{t-1} @ W_hh_slice^T : K = 256 in 8 WMMA k-steps ----
    #pragma unroll
    for (int kk = 0; kk < 8; ++kk) {
      const __bf16* ap = Hs + (w * 16 + l16) * 256 + kk * 32 + hi * 16;
      v8bf alo = *(const v8bf*)(ap);
      v8bf ahi = *(const v8bf*)(ap + 8);
      v16bf afrag = __builtin_shufflevector(alo, ahi,
          0, 1, 2, 3, 4, 5, 6, 7, 8, 9, 10, 11, 12, 13, 14, 15);
      #pragma unroll
      for (int g = 0; g < 4; ++g) {
        const __bf16* bp = Ws + (g * 16 + l16) * 256 + kk * 32 + hi * 16;
        v16bf bfrag = *(const v16bf*)(bp);
        acc[g] = wmma_bf16(afrag, bfrag, acc[g]);
      }
    }

    // ---- Pointwise LSTM cell (fp32) ----
    __bf16* hb_out = hb + (size_t)(t & 1) * (BATCH * HDIM);
    #pragma unroll
    for (int r = 0; r < 8; ++r) {
      const float ig = sigmoidf_fast(acc[0][r]);
      const float fg = sigmoidf_fast(acc[1][r]);
      const float gg = tanhf(acc[2][r]);
      const float og = sigmoidf_fast(acc[3][r]);
      const float cn = fg * cfrag[r] + ig * gg;
      const float hn = og * tanhf(cn);
      cfrag[r] = cn;
      const int m   = w * 16 + r + hi * 8;
      const int col = hbase + l16;
      out[(size_t)t * BATCH * HDIM + (size_t)m * HDIM + col] = hn;
      hb_out[(size_t)m * HDIM + col] = (__bf16)hn;       // feeds next step
      if (t == T_STEPS - 1)
        out[(size_t)T_STEPS * BATCH * HDIM + (size_t)m * HDIM + col] = hn; // h_T
    }

    // Prefetch next step's x-gate slice while we wait at the barrier
    if (t + 1 < T_STEPS) {
      const int m0 = w * 16 + hi * 8;
      __builtin_prefetch(xg + (size_t)((t + 1) * BATCH + m0) * GDIM + hbase + l16, 0, 1);
    }

    // ---- Grid-wide barrier (monotonic counter) ----
    __syncthreads();
    if (tid == 0) {
      __threadfence();
      atomicAdd(counter, 1u);
      const unsigned int need = (unsigned int)(gridDim.x) * (unsigned int)(t + 1);
      while (__hip_atomic_load(counter, __ATOMIC_RELAXED, __HIP_MEMORY_SCOPE_AGENT) < need) {
        __builtin_amdgcn_s_sleep(1);
      }
      __threadfence();
    }
    __syncthreads();
  }

  // c_T
  #pragma unroll
  for (int r = 0; r < 8; ++r) {
    const int m = w * 16 + r + hi * 8;
    out[(size_t)T_STEPS * BATCH * HDIM + (size_t)BATCH * HDIM +
        (size_t)m * HDIM + hbase + l16] = cfrag[r];
  }
}

__global__ void init_counter(unsigned int* c) {
  if (threadIdx.x == 0) *c = 0u;
}

// ---------------------------------------------------------------------------
extern "C" void kernel_launch(void* const* d_in, const int* in_sizes, int n_in,
                              void* d_out, int out_size, void* d_ws, size_t ws_size,
                              hipStream_t stream) {
  const float* x   = (const float*)d_in[0];  // [T,B,I]
  const float* h0  = (const float*)d_in[1];  // [B,H]
  const float* c0  = (const float*)d_in[2];  // [B,H]
  const float* Wih = (const float*)d_in[3];  // [4H,I]
  const float* Whh = (const float*)d_in[4];  // [4H,H]
  const float* bih = (const float*)d_in[5];  // [4H]
  const float* bhh = (const float*)d_in[6];  // [4H]
  float* out = (float*)d_out;

  // ws layout:
  //   [0, 4K)        barrier counter
  //   [4K, 4K+64K)   bf16 h ping-pong buffers (2 x 32 KB)
  //   [128K, ...)    XG fp32 [T*B, 4H] (512 MB)
  unsigned int* counter = (unsigned int*)d_ws;
  __bf16* hb = (__bf16*)((char*)d_ws + 4096);
  float* xg  = (float*)((char*)d_ws + 131072);

  init_counter<<<1, 64, 0, stream>>>(counter);

  dim3 g1(T_STEPS * BATCH / 128, GDIM / 128);   // (1024, 8)
  xgates_gemm<<<g1, 256, 0, stream>>>(x, Wih, bih, bhh, xg);

  // 16 persistent WGs, 128 threads, 128 KB dynamic LDS
  lstm_scan<<<16, 128, LDS_TOTAL, stream>>>(h0, c0, Whh, xg, out, hb, counter);
}
